// lgc_loss_79628693667849
// MI455X (gfx1250) — matrix operand order repaired and checked by reading, not compiled, not run on hardware
//
#include <hip/hip_runtime.h>
#include <hip/hip_bf16.h>
#include <math.h>

// ---------------------------------------------------------------------------
// Problem constants (from reference)
// ---------------------------------------------------------------------------
#define NPTS   32768          // N rows of z
#define KCLU   1024           // K rows of mu (columns of d_)
#define DH     512            // feature dim
#define EPSV   1e-5f
#define KNN    5
#define TGT    1.5f
#define NITERS 150            // K_NN * 30
#define BIGF   3.0e38f

typedef __bf16 bf16_t;
typedef __bf16 v16bf __attribute__((ext_vector_type(16)));
typedef float  v8f   __attribute__((ext_vector_type(8)));
typedef unsigned int u32x4 __attribute__((ext_vector_type(4)));
typedef unsigned int u32x4v __attribute__((ext_vector_type(4)));
typedef int i32x4v __attribute__((ext_vector_type(4)));
typedef int i32x8v __attribute__((ext_vector_type(8)));

#if __has_builtin(__builtin_amdgcn_tensor_load_to_lds)
#define HAVE_TDM 1
#else
#define HAVE_TDM 0
#endif

union FragU { v16bf v; u32x4 u[2]; };

static __device__ __forceinline__ v8f wmma_bf16(v16bf a, v16bf b, v8f c) {
  // 8 args: (neg_a, A, neg_b, B, c_mod, C, reuse_a, reuse_b)
  return __builtin_amdgcn_wmma_f32_16x16x32_bf16(false, a, false, b, (short)0, c,
                                                 false, false);
}

static __device__ __forceinline__ float waveRedSum(float v) {
#pragma unroll
  for (int off = 16; off > 0; off >>= 1) v += __shfl_xor(v, off, 32);
  return v;
}
static __device__ __forceinline__ float waveRedMin(float v) {
#pragma unroll
  for (int off = 16; off > 0; off >>= 1) v = fminf(v, __shfl_xor(v, off, 32));
  return v;
}

static __device__ __forceinline__ void wait_tensorcnt0() {
#if __has_builtin(__builtin_amdgcn_s_wait_tensorcnt)
  __builtin_amdgcn_s_wait_tensorcnt(0);
#else
  asm volatile("s_wait_tensorcnt 0x0" ::: "memory");
#endif
}

#if HAVE_TDM
// TDM: stage a [tile_dim1=128 rows] x [tile_dim0=32 elems] bf16 tile,
// row stride = 512 elems, into LDS contiguously (row-major, 64B per row).
// D# layout per CDNA5 ISA.
static __device__ __forceinline__ void tdm_load_tile(const void* gsrc,
                                                     unsigned ldsOff) {
  unsigned long long ga = (unsigned long long)(uintptr_t)gsrc;
  u32x4v g0 = {0u, 0u, 0u, 0u};
  g0[0] = 1u;                                          // count=1, user mode
  g0[1] = ldsOff;                                      // lds_addr (bytes)
  g0[2] = (unsigned)ga;                                // global_addr[31:0]
  g0[3] = (unsigned)((ga >> 32) & 0x01ffffffu) | (2u << 30);  // addr[56:32]|type=2
  i32x8v g1 = {0, 0, 0, 0, 0, 0, 0, 0};
  g1[0] = 0x00010000;                                  // data_size = 2B
  g1[1] = (int)(512u << 16);                           // tensor_dim0 = 512
  g1[2] = (int)(1024u << 16);                          // tensor_dim1 = 1024
  g1[3] = (int)(32u << 16);                            // tile_dim0   = 32
  g1[4] = 128;                                         // tile_dim1   = 128
  g1[5] = 512;                                         // tensor_dim0_stride = 512
  i32x4v z4 = {0, 0, 0, 0};
#if __clang_major__ >= 23
  i32x8v z8 = {0, 0, 0, 0, 0, 0, 0, 0};
  __builtin_amdgcn_tensor_load_to_lds(g0, g1, z4, z4, z8, 0);
#else
  __builtin_amdgcn_tensor_load_to_lds(g0, g1, z4, z4, 0);
#endif
}
#endif

// ---------------------------------------------------------------------------
// K0: split f32 -> bf16 hi/lo; per-row sum and sum-of-squares
// ---------------------------------------------------------------------------
__global__ __launch_bounds__(256) void prep_kernel(
    const float* __restrict__ src, bf16_t* __restrict__ hi, bf16_t* __restrict__ lo,
    float* __restrict__ sumsq, float* __restrict__ sums) {
  const int row = blockIdx.x;
  const int tid = threadIdx.x;
  __shared__ float sA[256];
  __shared__ float sB[256];
  float pss = 0.f, ps = 0.f;
  const size_t base = (size_t)row * DH;
  for (int c = tid; c < DH; c += 256) {
    float x  = src[base + c];
    bf16_t h = (bf16_t)x;
    float hr = (float)h;
    bf16_t l = (bf16_t)(x - hr);
    hi[base + c] = h;
    lo[base + c] = l;
    pss += x * x;
    ps  += x;
  }
  sA[tid] = pss; sB[tid] = ps;
  __syncthreads();
  for (int s = 128; s > 0; s >>= 1) {
    if (tid < s) { sA[tid] += sA[tid + s]; sB[tid] += sB[tid + s]; }
    __syncthreads();
  }
  if (tid == 0) { sumsq[row] = sA[0]; sums[row] = sB[0]; }
}

// ---------------------------------------------------------------------------
// K1: d_[i,j] = sqrt(max(zz+mm-2*z.mu + 2eps(sz-sm) + DH*eps^2, 0))
// split-bf16 compensated GEMM via v_wmma_f32_16x16x32_bf16.
// 256 threads = 8 waves; macro tile 128x128; wave tile 32x64 (2x4 WMMA tiles).
// B (mu) tile double-buffered in LDS via TENSOR_LOAD_TO_LDS (TDM):
// wave0 DMAs the hi tile, wave1 the lo tile (parallel descriptors).
// ---------------------------------------------------------------------------
__global__ __launch_bounds__(256) void gemm_dist_kernel(
    const bf16_t* __restrict__ zhi, const bf16_t* __restrict__ zlo,
    const bf16_t* __restrict__ mhi, const bf16_t* __restrict__ mlo,
    const float* __restrict__ zz, const float* __restrict__ sz,
    const float* __restrict__ mm, const float* __restrict__ sm,
    float* __restrict__ dout) {
  const int wv   = threadIdx.x >> 5;
  const int lane = threadIdx.x & 31;
  const int sub  = lane & 15;     // lane within half-wave
  const int hlf  = lane >> 4;     // which K-half this lane holds (ISA 16-bit layout)

  const int rowBase  = blockIdx.y * 128 + (wv & 3) * 32;  // 4 row strips of 32
  const int colStrip = (wv >> 2) * 64;                    // 2 col strips of 64
  const int gcolBase = blockIdx.x * 128;

  // [buffer][hi/lo][col*32 + k] ; 32KB total
  __shared__ bf16_t ldsB[2][2][128 * 32];

  v8f acc[2][4] = {};

  // A base pointers: k-offsets become load immediates after full unroll
  const bf16_t* pAh[2];
  const bf16_t* pAl[2];
#pragma unroll
  for (int rf = 0; rf < 2; ++rf) {
    const size_t ab = (size_t)(rowBase + rf * 16 + sub) * DH + hlf * 8;
    pAh[rf] = zhi + ab;
    pAl[rf] = zlo + ab;
  }

  // ---- stage k-step 0 into buffer 0 ----
#if HAVE_TDM
  if (wv < 2) {
    const bf16_t* src = (wv == 0) ? mhi : mlo;
    tdm_load_tile(src + (size_t)gcolBase * DH,
                  (unsigned)(uintptr_t)(const void*)&ldsB[0][wv][0]);
  }
#else
  {
    const int col = threadIdx.x >> 1, kh = threadIdx.x & 1;
    const size_t g = (size_t)(gcolBase + col) * DH + kh * 16;
    bf16_t* dH = &ldsB[0][0][col * 32 + kh * 16];
    bf16_t* dL = &ldsB[0][1][col * 32 + kh * 16];
    *(u32x4*)(dH)     = *(const u32x4*)(mhi + g);
    *(u32x4*)(dH + 8) = *(const u32x4*)(mhi + g + 8);
    *(u32x4*)(dL)     = *(const u32x4*)(mlo + g);
    *(u32x4*)(dL + 8) = *(const u32x4*)(mlo + g + 8);
  }
#endif

#pragma unroll
  for (int ks = 0; ks < DH / 32; ++ks) {
    const int buf = ks & 1;
    const int k0  = ks * 32;
#if HAVE_TDM
    if (wv < 2) wait_tensorcnt0();
#endif
    // One barrier proves: (a) DMA into buf landed (issuers waited tensorcnt),
    // (b) everyone finished reading buf^1 last iteration -> safe to refill it.
    __syncthreads();
    if (ks + 1 < DH / 32) {
#if HAVE_TDM
      if (wv < 2) {
        const bf16_t* src = (wv == 0) ? mhi : mlo;
        tdm_load_tile(src + (size_t)gcolBase * DH + (ks + 1) * 32,
                      (unsigned)(uintptr_t)(const void*)&ldsB[buf ^ 1][wv][0]);
      }
#else
      const int col = threadIdx.x >> 1, kh = threadIdx.x & 1;
      const size_t g = (size_t)(gcolBase + col) * DH + (ks + 1) * 32 + kh * 16;
      bf16_t* dH = &ldsB[buf ^ 1][0][col * 32 + kh * 16];
      bf16_t* dL = &ldsB[buf ^ 1][1][col * 32 + kh * 16];
      *(u32x4*)(dH)     = *(const u32x4*)(mhi + g);
      *(u32x4*)(dH + 8) = *(const u32x4*)(mhi + g + 8);
      *(u32x4*)(dL)     = *(const u32x4*)(mlo + g);
      *(u32x4*)(dL + 8) = *(const u32x4*)(mlo + g + 8);
#endif
    }

    // A fragments (global, immediate offsets): lanes 0-15 K={0..7,16..23},
    // lanes 16-31 K={8..15,24..31} (ISA 16-bit A layout)
    FragU ah[2], al[2];
#pragma unroll
    for (int rf = 0; rf < 2; ++rf) {
      ah[rf].u[0] = *(const u32x4*)(pAh[rf] + k0);
      ah[rf].u[1] = *(const u32x4*)(pAh[rf] + k0 + 16);
      al[rf].u[0] = *(const u32x4*)(pAl[rf] + k0);
      al[rf].u[1] = *(const u32x4*)(pAl[rf] + k0 + 16);
    }
#pragma unroll
    for (int cf = 0; cf < 4; ++cf) {
      // B fragment from LDS: lanes 0-15 K={0..7,16..23}, lanes 16-31 K={8..15,24..31} of column n
      const int cl = colStrip + cf * 16 + sub;
      const bf16_t* bH = &ldsB[buf][0][cl * 32 + hlf * 8];
      const bf16_t* bL = &ldsB[buf][1][cl * 32 + hlf * 8];
      FragU bh, bl;
      bh.u[0] = *(const u32x4*)(bH);
      bh.u[1] = *(const u32x4*)(bH + 16);
      bl.u[0] = *(const u32x4*)(bL);
      bl.u[1] = *(const u32x4*)(bL + 16);
#pragma unroll
      for (int rf = 0; rf < 2; ++rf) {
        acc[rf][cf] = wmma_bf16(ah[rf].v, bh.v, acc[rf][cf]);  // hi*hi
        acc[rf][cf] = wmma_bf16(ah[rf].v, bl.v, acc[rf][cf]);  // hi*lo
        acc[rf][cf] = wmma_bf16(al[rf].v, bh.v, acc[rf][cf]);  // lo*hi
      }
    }
  }

  // Epilogue: C/D layout — VGPR r, lanes 0-15: (M=r,N=lane); lanes 16-31: (M=r+8,N=lane-16)
  // d_ is stored with default (RT) policy: it is re-read twice and, at 128MB,
  // is deliberately kept resident in the 192MB L2.
  const float cterm = (float)DH * EPSV * EPSV;
#pragma unroll
  for (int cf = 0; cf < 4; ++cf) {
    const int col  = gcolBase + colStrip + cf * 16 + sub;
    const float mmv = mm[col];
    const float smv = sm[col];
#pragma unroll
    for (int rf = 0; rf < 2; ++rf) {
      const int rowb = rowBase + rf * 16 + (hlf ? 8 : 0);
      float* pd = dout + (size_t)rowb * KCLU + col;     // row stores: imm offsets
      const float* pzz = zz + rowb;
      const float* psz = sz + rowb;
#pragma unroll
      for (int r = 0; r < 8; ++r) {
        const float a  = acc[rf][cf][r];
        float d2 = pzz[r] + mmv - 2.f * a + 2.f * EPSV * (psz[r] - smv) + cterm;
        pd[(size_t)r * KCLU] = sqrtf(fmaxf(d2, 0.f));
      }
    }
  }
}

// ---------------------------------------------------------------------------
// K2: per-row 5 smallest of d_ (row length 1024). One wave per row.
// d_ re-read later -> regular (RT) loads, served from L2.
// ---------------------------------------------------------------------------
__global__ __launch_bounds__(256) void top5_rows_kernel(
    const float* __restrict__ d, float* __restrict__ rel) {
  const int wv   = threadIdx.x >> 5;
  const int lane = threadIdx.x & 31;
  const int row  = blockIdx.x * 8 + wv;
  const float* r = d + (size_t)row * KCLU;

  float t[5] = {BIGF, BIGF, BIGF, BIGF, BIGF};
#pragma unroll 4
  for (int it = 0; it < 32; ++it) {
    float v = r[lane + 32 * it];
    if (v < t[4]) {
      t[4] = v;
#pragma unroll
      for (int q = 4; q > 0; --q) {
        if (t[q] < t[q - 1]) { float tmp = t[q]; t[q] = t[q - 1]; t[q - 1] = tmp; }
      }
    }
  }
  // merge sorted per-lane lists: 5 rounds of wave arg-min
  int p = 0;
#pragma unroll
  for (int rnd = 0; rnd < KNN; ++rnd) {
    float cand = (p == 0) ? t[0] : (p == 1) ? t[1] : (p == 2) ? t[2]
               : (p == 3) ? t[3] : (p == 4) ? t[4] : BIGF;
    float m = waveRedMin(cand);
    unsigned long long msk = __ballot(cand == m);
    int src = __ffsll(msk) - 1;
    if (lane == src) p++;
    if (lane == 0) rel[(size_t)row * KNN + rnd] = m;
  }
}

// ---------------------------------------------------------------------------
// K3: per-row sigma via 150-step bisection (exact replication of the scan)
// ---------------------------------------------------------------------------
__global__ __launch_bounds__(256) void sigma_kernel(
    const float* __restrict__ rel, float* __restrict__ sigma) {
  const int i = blockIdx.x * 256 + threadIdx.x;
  if (i >= NPTS) return;
  float rv[KNN];
#pragma unroll
  for (int k = 0; k < KNN; ++k) rv[k] = rel[(size_t)i * KNN + k];
  const float rho = rv[0];
  float lo = 0.f, hi = 10000.f, sg = 1.f;
  for (int it = 0; it < NITERS; ++it) {
    float inv = 1.f / sg;
    float cur = 0.f;
#pragma unroll
    for (int k = 0; k < KNN; ++k) cur += __expf(-fmaxf(rv[k] - rho, 0.f) * inv);
    if (cur > TGT) hi = sg; else lo = sg;      // mid[idx] = sigma
    sg = 0.5f * (lo + hi);                     // sigma = mean(mid)
  }
  sigma[i] = sg;
}

// ---------------------------------------------------------------------------
// K4a: W1 = exp(-relu(d-rho)/sigma); S = W1/rowsum; partial colsums of S.
// 8 waves/block, each wave owns 16 rows + its own 1024-entry LDS colsum slice.
// Cache policy: d_ loads are last-use -> NT; W1 stores are write-only -> NT
// (don't evict L2-resident working set); S stores stay RT (re-read by dmat).
// ---------------------------------------------------------------------------
__global__ __launch_bounds__(256) void w1s_kernel(
    const float* __restrict__ d, const float* __restrict__ rel,
    const float* __restrict__ sigma,
    float* __restrict__ outW1a, float* __restrict__ outS,
    float* __restrict__ outW1b, float* __restrict__ partial) {
  __shared__ float cs[8 * 1024];   // 32 KB of the 320 KB WGP LDS
  const int tid  = threadIdx.x;
  const int wv   = tid >> 5;
  const int lane = tid & 31;
  for (int i = tid; i < 8 * 1024; i += 256) cs[i] = 0.f;
  __syncthreads();

  const int rowBase = blockIdx.x * 128 + wv * 16;
  float* mycs = cs + wv * 1024;

  for (int rr = 0; rr < 16; ++rr) {
    const int row   = rowBase + rr;
    const float rho = rel[(size_t)row * KNN];
    const float isg = 1.f / sigma[row];
    float w[32];
    float rs = 0.f;
#pragma unroll
    for (int t2 = 0; t2 < 32; ++t2) {
      const int col = lane + 32 * t2;
      float dv = __builtin_nontemporal_load(&d[(size_t)row * KCLU + col]);
      float wx = __expf(-fmaxf(dv - rho, 0.f) * isg);
      w[t2] = wx;
      rs += wx;
    }
    rs = waveRedSum(rs);
    const float invr = 1.f / rs;
#pragma unroll
    for (int t2 = 0; t2 < 32; ++t2) {
      const int col = lane + 32 * t2;
      const size_t idx = (size_t)row * KCLU + col;
      const float wx = w[t2];
      __builtin_nontemporal_store(wx, &outW1a[idx]);
      __builtin_nontemporal_store(wx, &outW1b[idx]);
      const float s = wx * invr;
      outS[idx] = s;
      mycs[col] += s;
    }
  }
  __syncthreads();
  for (int col = tid; col < 1024; col += 256) {
    float a = 0.f;
#pragma unroll
    for (int q = 0; q < 8; ++q) a += cs[q * 1024 + col];
    partial[(size_t)blockIdx.x * KCLU + col] = a;
  }
}

// K4b: final colsum C[j] = sum over 256 block-partials (deterministic order)
__global__ __launch_bounds__(256) void colsum_kernel(
    const float* __restrict__ partial, float* __restrict__ C) {
  const int j = blockIdx.x * 256 + threadIdx.x;
  float a = 0.f;
  for (int b = 0; b < 256; ++b) a += partial[(size_t)b * KCLU + j];
  C[j] = a;
}

// K4c: D = (S*S / C[j]) row-normalized. One wave per row.
// S loads are last-use -> NT; D stores are write-only -> NT.
__global__ __launch_bounds__(256) void dmat_kernel(
    const float* __restrict__ S, const float* __restrict__ C,
    float* __restrict__ Dout) {
  const int wv   = threadIdx.x >> 5;
  const int lane = threadIdx.x & 31;
  const int row  = blockIdx.x * 8 + wv;
  float v[32];
  float ts = 0.f;
#pragma unroll
  for (int t2 = 0; t2 < 32; ++t2) {
    const int col = lane + 32 * t2;
    float s = __builtin_nontemporal_load(&S[(size_t)row * KCLU + col]);
    float x = s * s / C[col];
    v[t2] = x;
    ts += x;
  }
  ts = waveRedSum(ts);
  const float inv = 1.f / ts;
#pragma unroll
  for (int t2 = 0; t2 < 32; ++t2) {
    const int col = lane + 32 * t2;
    __builtin_nontemporal_store(v[t2] * inv, &Dout[(size_t)row * KCLU + col]);
  }
}

// ---------------------------------------------------------------------------
// Host-side orchestration
// ---------------------------------------------------------------------------
extern "C" void kernel_launch(void* const* d_in, const int* in_sizes, int n_in,
                              void* d_out, int out_size, void* d_ws, size_t ws_size,
                              hipStream_t stream) {
  (void)in_sizes; (void)n_in; (void)out_size; (void)ws_size;
  const float* z  = (const float*)d_in[0];
  const float* mu = (const float*)d_in[1];
  // d_in[2] = epoch (unused by the reference forward)

  float* out = (float*)d_out;
  const size_t NK = (size_t)NPTS * KCLU;
  float* outW1a = out;              // W1
  float* outS   = out + NK;         // S
  float* outW1b = out + 2 * NK;     // W1 (duplicate output)
  float* outD   = out + 3 * NK;     // D  (also reused as d_ scratch until last pass)

  // workspace carve-out (256B aligned)
  char* ws = (char*)d_ws;
  size_t off = 0;
  auto carve = [&](size_t bytes) -> void* {
    void* p = ws + off;
    off += (bytes + 255) & ~(size_t)255;
    return p;
  };
  bf16_t* zhi = (bf16_t*)carve((size_t)NPTS * DH * sizeof(bf16_t));
  bf16_t* zlo = (bf16_t*)carve((size_t)NPTS * DH * sizeof(bf16_t));
  bf16_t* mhi = (bf16_t*)carve((size_t)KCLU * DH * sizeof(bf16_t));
  bf16_t* mlo = (bf16_t*)carve((size_t)KCLU * DH * sizeof(bf16_t));
  float* zz    = (float*)carve((size_t)NPTS * sizeof(float));
  float* szv   = (float*)carve((size_t)NPTS * sizeof(float));
  float* mm    = (float*)carve((size_t)KCLU * sizeof(float));
  float* smv   = (float*)carve((size_t)KCLU * sizeof(float));
  float* rel   = (float*)carve((size_t)NPTS * KNN * sizeof(float));
  float* sig   = (float*)carve((size_t)NPTS * sizeof(float));
  float* part  = (float*)carve((size_t)256 * KCLU * sizeof(float));
  float* csum  = (float*)carve((size_t)KCLU * sizeof(float));

  float* dmatS = outD;  // d_ lives in the D output slot until dmat_kernel

  // K0: bf16 hi/lo split + row reductions
  prep_kernel<<<NPTS, 256, 0, stream>>>(z, zhi, zlo, zz, szv);
  prep_kernel<<<KCLU, 256, 0, stream>>>(mu, mhi, mlo, mm, smv);

  // K1: WMMA GEMM (TDM-staged B tiles) + fused distance epilogue
  dim3 ggrid(KCLU / 128, NPTS / 128);
  gemm_dist_kernel<<<ggrid, 256, 0, stream>>>(zhi, zlo, mhi, mlo,
                                              zz, szv, mm, smv, dmatS);

  // K2: per-row top-5 (rho_z = rel[:,0]); column-side (rho_u/sigma_u/W2) is
  // dead code in the reference — skipped entirely.
  top5_rows_kernel<<<NPTS / 8, 256, 0, stream>>>(dmatS, rel);

  // K3: sigma bisection
  sigma_kernel<<<NPTS / 256, 256, 0, stream>>>(rel, sig);

  // K4: W1 / S / colsum / D
  w1s_kernel<<<NPTS / 128, 256, 0, stream>>>(dmatS, rel, sig,
                                             outW1a, outS, outW1b, part);
  colsum_kernel<<<KCLU / 256, 256, 0, stream>>>(part, csum);
  dmat_kernel<<<NPTS / 8, 256, 0, stream>>>(outS, csum, outD);
}
